// Processor_2241972929028
// MI455X (gfx1250) — matrix operand order repaired and checked by reading, not compiled, not run on hardware
//
#include <hip/hip_runtime.h>
#include <hip/hip_bf16.h>

// ---------------------------------------------------------------------------
// GNN message-passing for MI455X (gfx1250, wave32).
// Edge/node MLP GEMMs run on v_wmma_f32_16x16x32_bf16 (fp32 accumulate).
// M-tile = 64 rows per workgroup: halves L2 weight traffic vs M=32 and gives
// 4 WMMAs per A-fragment ds_load. LDS = exactly 64 KB via buffer aliasing.
// Weights are pre-transposed + converted to bf16 once into workspace so each
// B fragment is 16 contiguous bf16 per lane (two global_load_b128 from L2).
// Segment-sum via global_atomic_add_f32 into an L2-resident agg buffer.
// ---------------------------------------------------------------------------

#define DEV_INLINE __device__ __forceinline__

typedef __attribute__((ext_vector_type(16))) __bf16 bf16x16;
typedef __attribute__((ext_vector_type(8)))  __bf16 bf16x8;
typedef __attribute__((ext_vector_type(4)))  __bf16 bf16x4;
typedef __attribute__((ext_vector_type(8)))  float  f32x8;

static_assert(sizeof(__bf16) == 2, "bf16 storage");

constexpr int       Bdim      = 2;
constexpr int       Nn        = 10000;
constexpr int       Dd        = 128;
constexpr long long Ee        = 320000;
constexpr int       NLAYERS   = 10;
constexpr float     EPSV      = 1e-5f;
constexpr int       NODE_ROWS = Bdim * Nn;   // 20000

// ---------------------------------------------------------------------------
// WMMA tiled GEMM: X (LDS, 64 x K bf16 row-major)  @  Wt^T (global, 128 x K
// bf16, row n = output column n)  + bias  ->  64 x 128.
// 8 waves: wave w owns m-tile (w>>1) and n-tiles (w&1)*4 .. +3.
// A fragment (contiguous k order, matching B): lane half = lane>>4, row = lane&15,
//   elems 0-7  -> K = kt*32 + half*16 + 0..7   (one ds_load_b128)
//   elems 8-15 -> K = kt*32 + half*16 + 8..15  (one ds_load_b128)
// B fragment: lane col = lane&15, K = kt*32 + half*16 + 0..15 contiguous.
// ---------------------------------------------------------------------------
template <int K, bool RELU, bool F32OUT>
DEV_INLINE void mlp_gemm64(const __bf16* __restrict__ X,
                           const __bf16* __restrict__ Wt,
                           const float*  __restrict__ bias,
                           __bf16* __restrict__ Hout,   // LDS 64x128 (bf16 path)
                           float*  __restrict__ Fout)   // LDS 64x128 (f32 path)
{
    const int lane  = threadIdx.x & 31;
    const int wave  = threadIdx.x >> 5;
    const int mtile = wave >> 1;          // 0..3
    const int nb    = (wave & 1) * 4;     // n-tile base: 0 or 4
    const int half  = lane >> 4;          // 0/1
    const int lidx  = lane & 15;

    f32x8 acc[4] = {};

    const __bf16* arowbase = X + (mtile * 16 + lidx) * K + half * 16;

    for (int kt = 0; kt < K / 32; ++kt) {
        bf16x8 alo = *(const bf16x8*)(arowbase + kt * 32);
        bf16x8 ahi = *(const bf16x8*)(arowbase + kt * 32 + 8);
        bf16x16 a;
        #pragma unroll
        for (int i = 0; i < 8; ++i) { a[i] = alo[i]; a[i + 8] = ahi[i]; }

        #pragma unroll
        for (int t = 0; t < 4; ++t) {
            const __bf16* brow = Wt + ((nb + t) * 16 + lidx) * K + kt * 32 + half * 16;
            bf16x8 blo = *(const bf16x8*)(brow);
            bf16x8 bhi = *(const bf16x8*)(brow + 8);
            bf16x16 b;
            #pragma unroll
            for (int i = 0; i < 8; ++i) { b[i] = blo[i]; b[i + 8] = bhi[i]; }

            acc[t] = __builtin_amdgcn_wmma_f32_16x16x32_bf16(
                false, a, false, b, (short)0, acc[t], false, false);
        }
    }

    // Epilogue: C/D layout -> row m = mtile*16 + half*8 + r, col n.
    #pragma unroll
    for (int t = 0; t < 4; ++t) {
        const int   n  = (nb + t) * 16 + lidx;
        const float bv = bias[n];
        #pragma unroll
        for (int r = 0; r < 8; ++r) {
            const int m = mtile * 16 + half * 8 + r;
            float v = acc[t][r] + bv;
            if (RELU) v = fmaxf(v, 0.f);
            if (F32OUT) Fout[m * 128 + n] = v;
            else        Hout[m * 128 + n] = (__bf16)v;
        }
    }
}

// Convert 32 fp32 -> 32 bf16 into LDS (eight ds_store_b64).
DEV_INLINE void stage32(__bf16* __restrict__ dst, const float* __restrict__ src)
{
    #pragma unroll
    for (int i = 0; i < 32; i += 4) {
        float4 v = *(const float4*)(src + i);
        bf16x4 o;
        o[0] = (__bf16)v.x; o[1] = (__bf16)v.y; o[2] = (__bf16)v.z; o[3] = (__bf16)v.w;
        *(bf16x4*)(dst + i) = o;
    }
}

DEV_INLINE void stage32_zero(__bf16* __restrict__ dst)
{
    bf16x4 z = {};
    #pragma unroll
    for (int i = 0; i < 32; i += 4) *(bf16x4*)(dst + i) = z;
}

// ---------------------------------------------------------------------------
// Edge layer: gather -> 3-layer MLP (WMMA) -> LayerNorm -> residual + scatter.
// One block = 64 edges. E/64 = 5000 blocks, no tail.
// LDS (64 KB): sX 48 KB | sH1 16 KB; sH2 aliases sX[0:16K], sF sX[16K:48K].
// ---------------------------------------------------------------------------
__global__ __launch_bounds__(256) void edge_layer_kernel(
    const float* __restrict__ nodes, float* __restrict__ edges,
    const long long* __restrict__ nbr,
    const __bf16* __restrict__ Wt0, const float* __restrict__ b0,
    const __bf16* __restrict__ Wt1, const float* __restrict__ b1,
    const __bf16* __restrict__ Wt2, const float* __restrict__ b2,
    const float* __restrict__ gamma, const float* __restrict__ beta,
    float* __restrict__ agg)
{
    __shared__ __align__(16) __bf16 sX [64 * 384];   // 48 KB
    __shared__ __align__(16) __bf16 sH1[64 * 128];   // 16 KB
    __bf16* sH2 = sX;                                // 16 KB alias (X dead after GEMM1)
    float*  sF  = (float*)(sX + 64 * 128);           // 32 KB alias (bytes 16K..48K of sX)

    const long long e0 = (long long)blockIdx.x * 64;
    const int t = threadIdx.x;

    // Warm L2 with this layer's weights (global_prefetch_b8).
    if (t < 128) {
        __builtin_prefetch(Wt0 + t * 384, 0, 1);
        __builtin_prefetch(Wt1 + t * 128, 0, 1);
        __builtin_prefetch(Wt2 + t * 128, 0, 1);
    }

    // ---- stage concat(edges, recv, send) as bf16: 4 threads per row ----
    {
        const int row = t >> 2;                 // 0..63
        const int q   = t & 3;                  // 32-col chunk
        const long long e  = e0 + row;
        const long long bb = nbr[e * 3 + 0];
        const long long rr = nbr[e * 3 + 1];
        const long long ss = nbr[e * 3 + 2];
        const float* esrc = edges + e * Dd;
        const float* rsrc = nodes + (bb * Nn + rr) * Dd;
        const float* ssrc = nodes + (bb * Nn + ss) * Dd;
        __bf16* xrow = sX + row * 384;
        stage32(xrow + 0   + q * 32, esrc + q * 32);
        stage32(xrow + 128 + q * 32, rsrc + q * 32);
        stage32(xrow + 256 + q * 32, ssrc + q * 32);
    }
    __syncthreads();

    mlp_gemm64<384, true,  false>(sX,  Wt0, b0, sH1, nullptr);
    __syncthreads();
    mlp_gemm64<128, true,  false>(sH1, Wt1, b1, sH2, nullptr);
    __syncthreads();
    mlp_gemm64<128, false, true >(sH2, Wt2, b2, nullptr, sF);
    __syncthreads();

    // ---- LayerNorm + residual + segment-sum scatter (one thread per row) ----
    if (t < 64) {
        const long long e = e0 + t;
        const float* frow = sF + t * 128;

        float mu = 0.f;
        for (int i = 0; i < 128; i += 4) {
            float4 v = *(const float4*)(frow + i);
            mu += v.x + v.y + v.z + v.w;
        }
        mu *= (1.f / 128.f);
        float sq = 0.f;
        for (int i = 0; i < 128; i += 4) {
            float4 v = *(const float4*)(frow + i);
            float a = v.x - mu, b = v.y - mu, c = v.z - mu, d = v.w - mu;
            sq += a * a + b * b + c * c + d * d;
        }
        const float rstd = rsqrtf(sq * (1.f / 128.f) + EPSV);

        const long long bb = nbr[e * 3 + 0];
        const long long rr = nbr[e * 3 + 1];
        float* aggrow = agg   + (bb * Nn + rr) * Dd;
        float* erow   = edges + e * Dd;

        for (int i = 0; i < 128; i += 4) {
            float4 f  = *(const float4*)(frow + i);
            float4 g  = *(const float4*)(gamma + i);
            float4 bt = *(const float4*)(beta + i);
            float4 eo;
            eo.x = (f.x - mu) * rstd * g.x + bt.x;
            eo.y = (f.y - mu) * rstd * g.y + bt.y;
            eo.z = (f.z - mu) * rstd * g.z + bt.z;
            eo.w = (f.w - mu) * rstd * g.w + bt.w;
            float4 ev = *(const float4*)(erow + i);
            ev.x += eo.x; ev.y += eo.y; ev.z += eo.z; ev.w += eo.w;
            *(float4*)(erow + i) = ev;
            __hip_atomic_fetch_add(aggrow + i + 0, eo.x, __ATOMIC_RELAXED, __HIP_MEMORY_SCOPE_AGENT);
            __hip_atomic_fetch_add(aggrow + i + 1, eo.y, __ATOMIC_RELAXED, __HIP_MEMORY_SCOPE_AGENT);
            __hip_atomic_fetch_add(aggrow + i + 2, eo.z, __ATOMIC_RELAXED, __HIP_MEMORY_SCOPE_AGENT);
            __hip_atomic_fetch_add(aggrow + i + 3, eo.w, __ATOMIC_RELAXED, __HIP_MEMORY_SCOPE_AGENT);
        }
    }
}

// ---------------------------------------------------------------------------
// Node layer: concat(agg, nodes) -> 3-layer MLP (WMMA) -> LayerNorm -> residual.
// One block = 64 node rows; 313 blocks, tail-guarded.
// LDS (64 KB): sX 32 KB | sH1 16 KB | sH2 16 KB; sF aliases sX.
// ---------------------------------------------------------------------------
__global__ __launch_bounds__(256) void node_layer_kernel(
    float* __restrict__ nodes, const float* __restrict__ agg,
    const __bf16* __restrict__ Wt0, const float* __restrict__ b0,
    const __bf16* __restrict__ Wt1, const float* __restrict__ b1,
    const __bf16* __restrict__ Wt2, const float* __restrict__ b2,
    const float* __restrict__ gamma, const float* __restrict__ beta)
{
    __shared__ __align__(16) __bf16 sX [64 * 256];   // 32 KB
    __shared__ __align__(16) __bf16 sH1[64 * 128];   // 16 KB
    __shared__ __align__(16) __bf16 sH2[64 * 128];   // 16 KB
    float* sF = (float*)sX;                          // 32 KB alias (X dead after GEMM1)

    const long long r0 = (long long)blockIdx.x * 64;
    const int t = threadIdx.x;

    if (t < 128) {
        __builtin_prefetch(Wt0 + t * 256, 0, 1);
        __builtin_prefetch(Wt1 + t * 128, 0, 1);
        __builtin_prefetch(Wt2 + t * 128, 0, 1);
    }

    {
        const int row = t >> 2;
        const int q   = t & 3;
        const long long g = r0 + row;
        __bf16* xrow = sX + row * 256;
        if (g < NODE_ROWS) {
            stage32(xrow + 0   + q * 32, agg   + g * Dd + q * 32);
            stage32(xrow + 128 + q * 32, nodes + g * Dd + q * 32);
        } else {
            stage32_zero(xrow + 0   + q * 32);
            stage32_zero(xrow + 128 + q * 32);
        }
    }
    __syncthreads();

    mlp_gemm64<256, true,  false>(sX,  Wt0, b0, sH1, nullptr);
    __syncthreads();
    mlp_gemm64<128, true,  false>(sH1, Wt1, b1, sH2, nullptr);
    __syncthreads();
    mlp_gemm64<128, false, true >(sH2, Wt2, b2, nullptr, sF);
    __syncthreads();

    if (t < 64 && (r0 + t) < NODE_ROWS) {
        const long long g = r0 + t;
        const float* frow = sF + t * 128;

        float mu = 0.f;
        for (int i = 0; i < 128; i += 4) {
            float4 v = *(const float4*)(frow + i);
            mu += v.x + v.y + v.z + v.w;
        }
        mu *= (1.f / 128.f);
        float sq = 0.f;
        for (int i = 0; i < 128; i += 4) {
            float4 v = *(const float4*)(frow + i);
            float a = v.x - mu, b = v.y - mu, c = v.z - mu, d = v.w - mu;
            sq += a * a + b * b + c * c + d * d;
        }
        const float rstd = rsqrtf(sq * (1.f / 128.f) + EPSV);

        float* nrow = nodes + g * Dd;
        for (int i = 0; i < 128; i += 4) {
            float4 f  = *(const float4*)(frow + i);
            float4 gm = *(const float4*)(gamma + i);
            float4 bt = *(const float4*)(beta + i);
            float4 nv = *(const float4*)(nrow + i);
            nv.x += (f.x - mu) * rstd * gm.x + bt.x;
            nv.y += (f.y - mu) * rstd * gm.y + bt.y;
            nv.z += (f.z - mu) * rstd * gm.z + bt.z;
            nv.w += (f.w - mu) * rstd * gm.w + bt.w;
            *(float4*)(nrow + i) = nv;
        }
    }
}

// ---------------------------------------------------------------------------
// Utility kernels
// ---------------------------------------------------------------------------
__global__ void copy_f32_kernel(const float* __restrict__ src, float* __restrict__ dst, long long n)
{
    long long i = (long long)blockIdx.x * blockDim.x + threadIdx.x;
    const long long stride = (long long)gridDim.x * blockDim.x;
    for (; i < n; i += stride) dst[i] = src[i];
}

__global__ void zero_f32_kernel(float* __restrict__ p, long long n)
{
    long long i = (long long)blockIdx.x * blockDim.x + threadIdx.x;
    const long long stride = (long long)gridDim.x * blockDim.x;
    for (; i < n; i += stride) p[i] = 0.f;
}

// W: K x Nout fp32 (row-major)  ->  Wt: Nout x K bf16 (row n = output col n).
__global__ void convert_wt_kernel(const float* __restrict__ W, __bf16* __restrict__ Wt,
                                  int K, int Nout)
{
    const int i = blockIdx.x * blockDim.x + threadIdx.x;
    if (i >= K * Nout) return;
    const int n = i / K;
    const int k = i - n * K;
    Wt[i] = (__bf16)W[(size_t)k * Nout + n];
}

__global__ void pack_out_kernel(const float* __restrict__ nodes,
                                const float* __restrict__ edges,
                                const long long* __restrict__ nbr,
                                float* __restrict__ out, long long n)
{
    const long long NS = (long long)NODE_ROWS * Dd;   // 2,560,000
    const long long ES = Ee * Dd;                     // 40,960,000
    long long i = (long long)blockIdx.x * blockDim.x + threadIdx.x;
    const long long stride = (long long)gridDim.x * blockDim.x;
    for (; i < n; i += stride) {
        if (i < NS)           out[i] = nodes[i];
        else if (i < NS + ES) out[i] = edges[i - NS];
        else                  out[i] = (float)nbr[i - NS - ES];
    }
}

// ---------------------------------------------------------------------------
// Host driver
// d_in order (dict insertion order, nested depth-first as written in reference):
//   [0] nodes, [1] edges,
//   per layer l (base = 2 + 16*l):
//     +0..5  edge_mlp  W0,b0,W1,b1,W2,b2
//     +6,+7  edge_ln   gamma,beta
//     +8..13 node_mlp  W0,b0,W1,b1,W2,b2
//     +14,15 node_ln   gamma,beta
//   [n_in-1] neighbor_idxs (int64, E x 3)
// ---------------------------------------------------------------------------
extern "C" void kernel_launch(void* const* d_in, const int* in_sizes, int n_in,
                              void* d_out, int out_size, void* d_ws, size_t ws_size,
                              hipStream_t stream)
{
    (void)in_sizes; (void)ws_size;

    const float*     nodes_in = (const float*)d_in[0];
    const float*     edges_in = (const float*)d_in[1];
    const long long* nbr      = (const long long*)d_in[n_in - 1];

    // ---- workspace carve-up ----
    char* ws = (char*)d_ws;
    size_t off = 0;
    float* edges_ws = (float*)(ws + off); off += (size_t)Ee * Dd * sizeof(float);        // 163.84 MB
    float* nodes_ws = (float*)(ws + off); off += (size_t)NODE_ROWS * Dd * sizeof(float); //  10.24 MB
    float* agg      = (float*)(ws + off); off += (size_t)NODE_ROWS * Dd * sizeof(float); //  10.24 MB
    __bf16* wtp = (__bf16*)(ws + off);

    const int wdims[6][2] = { {384,128}, {128,128}, {128,128},   // edge MLP
                              {256,128}, {128,128}, {128,128} }; // node MLP
    __bf16* wt[NLAYERS][6];
    for (int l = 0; l < NLAYERS; ++l)
        for (int j = 0; j < 6; ++j) {
            wt[l][j] = wtp;
            wtp += (size_t)wdims[j][0] * wdims[j][1];
        }

    // ---- one-time prep: copy state, transpose+convert weights to bf16 ----
    copy_f32_kernel<<<4096, 256, 0, stream>>>(edges_in, edges_ws, Ee * Dd);
    copy_f32_kernel<<<2048, 256, 0, stream>>>(nodes_in, nodes_ws, (long long)NODE_ROWS * Dd);

    for (int l = 0; l < NLAYERS; ++l) {
        const int base = 2 + 16 * l;
        const int widx[6] = { base + 0, base + 2, base + 4, base + 8, base + 10, base + 12 };
        for (int j = 0; j < 6; ++j) {
            const int total = wdims[j][0] * wdims[j][1];
            convert_wt_kernel<<<(total + 255) / 256, 256, 0, stream>>>(
                (const float*)d_in[widx[j]], wt[l][j], wdims[j][0], wdims[j][1]);
        }
    }

    // ---- layer loop ----
    const int edge_blocks = (int)(Ee / 64);              // 5000
    const int node_blocks = (NODE_ROWS + 63) / 64;       // 313
    for (int l = 0; l < NLAYERS; ++l) {
        const int base = 2 + 16 * l;
        zero_f32_kernel<<<2048, 256, 0, stream>>>(agg, (long long)NODE_ROWS * Dd);

        edge_layer_kernel<<<edge_blocks, 256, 0, stream>>>(
            nodes_ws, edges_ws, nbr,
            wt[l][0], (const float*)d_in[base + 1],
            wt[l][1], (const float*)d_in[base + 3],
            wt[l][2], (const float*)d_in[base + 5],
            (const float*)d_in[base + 6], (const float*)d_in[base + 7],
            agg);

        node_layer_kernel<<<node_blocks, 256, 0, stream>>>(
            nodes_ws, agg,
            wt[l][3], (const float*)d_in[base + 9],
            wt[l][4], (const float*)d_in[base + 11],
            wt[l][5], (const float*)d_in[base + 13],
            (const float*)d_in[base + 14], (const float*)d_in[base + 15]);
    }

    // ---- pack outputs: nodes | edges | neighbor_idxs ----
    long long total = (long long)NODE_ROWS * Dd + Ee * Dd + Ee * 3;
    if ((long long)out_size < total) total = out_size;
    pack_out_kernel<<<4096, 256, 0, stream>>>(nodes_ws, edges_ws, nbr, (float*)d_out, total);
}
